// MixtureOfExperts_56925496541299
// MI455X (gfx1250) — compile-verified, matrix operand order checked by reading, not run on hardware
//
#include <hip/hip_runtime.h>
#include <math.h>

// ---------------- problem constants (from reference) ----------------
#define N_TOK  8192          // 4 * 2048
#define DDIM   1024
#define HDIM   4096
#define NEXP   4

typedef __attribute__((ext_vector_type(16))) __bf16 v16bf;
typedef __attribute__((ext_vector_type(8)))  float  v8f;

// workspace layout (bytes)
#define OFF_GATES 0u
#define SZ_GATES  (N_TOK * NEXP * 4)                          // 128 KB, f32 combine weights
#define OFF_XB    (OFF_GATES + SZ_GATES)
#define SZ_XB     (N_TOK * DDIM * 2)                          // 16 MB bf16
#define OFF_W1B   (OFF_XB + SZ_XB)
#define SZ_W1B    ((size_t)NEXP * HDIM * DDIM * 2)            // 32 MB bf16, panel layout
#define OFF_W2B   (OFF_W1B + SZ_W1B)
#define SZ_W2B    ((size_t)NEXP * DDIM * HDIM * 2)            // 32 MB bf16, panel layout

__device__ __forceinline__ __bf16 f2bf(float f) { return (__bf16)f; }

// -------------------------------------------------------------------
// 1) Gating: gates = x @ Wg + bg ; top-2 ; softmax over top-2.
//    One wave32 per token -> dense combine weights w[t][e].
// -------------------------------------------------------------------
__global__ void gate_kernel(const float* __restrict__ x,
                            const float* __restrict__ Wg,
                            const float* __restrict__ bg,
                            float* __restrict__ gates_out) {
  const int lane = threadIdx.x & 31;
  const int wave = threadIdx.x >> 5;
  const int t = blockIdx.x * (blockDim.x >> 5) + wave;
  if (t >= N_TOK) return;

  float g0 = 0.f, g1 = 0.f, g2 = 0.f, g3 = 0.f;
  const float*  xr  = x + (size_t)t * DDIM;
  const float4* wg4 = (const float4*)Wg;   // Wg is [D][4] row-major -> float4 per d
  for (int d = lane; d < DDIM; d += 32) {
    float  xv = xr[d];
    float4 w  = wg4[d];
    g0 += xv * w.x; g1 += xv * w.y; g2 += xv * w.z; g3 += xv * w.w;
  }
  for (int off = 16; off > 0; off >>= 1) {
    g0 += __shfl_xor(g0, off, 32);
    g1 += __shfl_xor(g1, off, 32);
    g2 += __shfl_xor(g2, off, 32);
    g3 += __shfl_xor(g3, off, 32);
  }
  if (lane == 0) {
    float g[NEXP] = {g0 + bg[0], g1 + bg[1], g2 + bg[2], g3 + bg[3]};
    int i0 = 0;
    #pragma unroll
    for (int e = 1; e < NEXP; ++e) if (g[e] > g[i0]) i0 = e;
    int i1 = (i0 == 0) ? 1 : 0;
    #pragma unroll
    for (int e = 0; e < NEXP; ++e) if (e != i0 && g[e] > g[i1]) i1 = e;
    float e1  = __expf(g[i1] - g[i0]);      // g[i0] >= g[i1]
    float inv = 1.f / (1.f + e1);
    float w[NEXP] = {0.f, 0.f, 0.f, 0.f};
    w[i0] = inv;
    w[i1] = e1 * inv;
    float4* dst = (float4*)(gates_out + (size_t)t * NEXP);
    *dst = make_float4(w[0], w[1], w[2], w[3]);
  }
}

// -------------------------------------------------------------------
// 2) Convert x -> bf16 (row-major, unchanged layout)
// -------------------------------------------------------------------
__global__ void cvt_x_kernel(const float* __restrict__ x, __bf16* __restrict__ xb) {
  const size_t total = (size_t)N_TOK * DDIM;
  for (size_t i = (size_t)blockIdx.x * blockDim.x + threadIdx.x; i < total;
       i += (size_t)gridDim.x * blockDim.x)
    xb[i] = f2bf(x[i]);
}

// -------------------------------------------------------------------
// 3) Repack W1 [E][D][H] f32 -> W1b panels: [E][H/16][16 col][D k] bf16
// -------------------------------------------------------------------
__global__ void repack_w1_kernel(const float* __restrict__ W1, __bf16* __restrict__ W1b) {
  const size_t total = (size_t)NEXP * DDIM * HDIM;
  for (size_t i = (size_t)blockIdx.x * blockDim.x + threadIdx.x; i < total;
       i += (size_t)gridDim.x * blockDim.x) {
    size_t e = i / ((size_t)DDIM * HDIM);
    size_t r = i % ((size_t)DDIM * HDIM);
    size_t d = r / HDIM;
    size_t h = r % HDIM;
    size_t dst = ((e * (HDIM / 16) + (h >> 4)) * 16 + (h & 15)) * DDIM + d;
    W1b[dst] = f2bf(W1[i]);
  }
}

// 4) Repack W2 [E][H][D] f32 -> W2b panels: [E][D/16][16 col][H k] bf16
__global__ void repack_w2_kernel(const float* __restrict__ W2, __bf16* __restrict__ W2b) {
  const size_t total = (size_t)NEXP * HDIM * DDIM;
  for (size_t i = (size_t)blockIdx.x * blockDim.x + threadIdx.x; i < total;
       i += (size_t)gridDim.x * blockDim.x) {
    size_t e = i / ((size_t)HDIM * DDIM);
    size_t r = i % ((size_t)HDIM * DDIM);
    size_t h = r / DDIM;
    size_t d = r % DDIM;
    size_t dst = ((e * (DDIM / 16) + (d >> 4)) * 16 + (d & 15)) * HDIM + h;
    W2b[dst] = f2bf(W2[i]);
  }
}

// -------------------------------------------------------------------
// 5) Fused MoE FFN main kernel, M = 32 tokens per block.
//    Every weight B-fragment (W1 and W2) feeds TWO WMMAs (the two
//    16-row M tiles) -> 32 FLOP per L2 byte of weight traffic.
// -------------------------------------------------------------------
__global__ void __launch_bounds__(256)
moe_main_kernel(const __bf16* __restrict__ xb,
                const __bf16* __restrict__ W1b,
                const __bf16* __restrict__ W2b,
                const float*  __restrict__ b1,
                const float*  __restrict__ b2,
                const float*  __restrict__ gates,
                float* __restrict__ out) {
  __shared__ __bf16 As[32 * DDIM];    // 64 KB: 32-token x-tile, row-major [m][k]
  __shared__ __bf16 Hs[32 * 128];     //  8 KB: stage-1 activations, row-major [m][h]
  __shared__ float  Wgate[32 * NEXP]; // 512 B

  const int tid   = threadIdx.x;
  const int lane  = tid & 31;
  const int wave  = tid >> 5;
  const int t0    = blockIdx.x * 32;      // first token of tile
  const int nloc  = lane & 15;            // N index inside 16x16 tile
  const int khalf = (lane >> 4) << 4;     // 0 or 16: K-half held by this lane
  const int mbase = (lane >> 4) * 8;      // C/D layout: VGPR r holds row mbase+r

  // --- cooperative loads into LDS ---
  {
    const uint4* src = (const uint4*)(xb + (size_t)t0 * DDIM);
    uint4* dst = (uint4*)As;
    #pragma unroll
    for (int i = tid; i < (32 * DDIM * 2) / 16; i += 256) dst[i] = src[i];
    if (tid < 32 * NEXP) Wgate[tid] = gates[(size_t)t0 * NEXP + tid];
  }
  __syncthreads();

  // persistent f32 accumulators: wave owns columns [wave*128, wave*128+128)
  // for both 16-row M tiles. acc[t*2+m2], t = n-tile, m2 = M-tile.
  v8f acc[16];
  #pragma unroll
  for (int t = 0; t < 16; ++t) acc[t] = (v8f)(0.f);

  const __bf16* arow0 = As + (size_t)(lane & 15) * DDIM + khalf;        // M tile 0
  const __bf16* arow1 = As + (size_t)(16 + (lane & 15)) * DDIM + khalf; // M tile 1

  for (int e = 0; e < NEXP; ++e) {
    for (int ch = 0; ch < HDIM / 128; ++ch) {
      const int hc = ch * 128;
      // ------------- stage 1: two 16x16 h tiles per wave (shared B) -------
      const int h0 = hc + wave * 16;
      v8f acc1a = (v8f)(0.f), acc1b = (v8f)(0.f);
      const __bf16* bpan =
          W1b + ((size_t)(e * (HDIM / 16) + (h0 >> 4)) * 16 + nloc) * DDIM + khalf;
      // prefetch next chunk's W1 panel (global_prefetch_b8)
      if (ch + 1 < HDIM / 128)
        __builtin_prefetch((const void*)(bpan + (size_t)128 * DDIM), 0, 1);
      #pragma unroll 8
      for (int k0 = 0; k0 < DDIM; k0 += 32) {
        v16bf b  = *(const v16bf*)(bpan + k0);
        v16bf a0 = *(const v16bf*)(arow0 + k0);
        v16bf a1 = *(const v16bf*)(arow1 + k0);
        acc1a = __builtin_amdgcn_wmma_f32_16x16x32_bf16(
            false, a0, false, b, (short)0, acc1a, false, false);
        acc1b = __builtin_amdgcn_wmma_f32_16x16x32_bf16(
            false, a1, false, b, (short)0, acc1b, false, false);
      }
      // bias + exact GELU + fold gate weight into h rows
      const float bias1 = b1[(size_t)e * HDIM + h0 + nloc];
      #pragma unroll
      for (int r = 0; r < 8; ++r) {
        const int m0 = mbase + r;
        float v = acc1a[r] + bias1;
        v = 0.5f * v * (1.f + erff(v * 0.70710678118654752f));
        v *= Wgate[m0 * NEXP + e];
        Hs[m0 * 128 + wave * 16 + nloc] = f2bf(v);
        const int m1 = 16 + m0;
        float u = acc1b[r] + bias1;
        u = 0.5f * u * (1.f + erff(u * 0.70710678118654752f));
        u *= Wgate[m1 * NEXP + e];
        Hs[m1 * 128 + wave * 16 + nloc] = f2bf(u);
      }
      __syncthreads();   // Hs chunk complete for all 8 waves

      // ------------- stage 2: acc += h_chunk @ W2 panel (shared B) --------
      const __bf16* hrow0 = Hs + (size_t)(lane & 15) * 128 + khalf;
      const __bf16* hrow1 = Hs + (size_t)(16 + (lane & 15)) * 128 + khalf;
      #pragma unroll
      for (int k0 = 0; k0 < 128; k0 += 32) {
        v16bf a0 = *(const v16bf*)(hrow0 + k0);
        v16bf a1 = *(const v16bf*)(hrow1 + k0);
        #pragma unroll
        for (int t = 0; t < 8; ++t) {
          const int dtile = wave * 8 + t;
          const __bf16* b2pan =
              W2b + ((size_t)(e * (DDIM / 16) + dtile) * 16 + nloc) * HDIM + hc + khalf;
          if (k0 == 0 && ch + 1 < HDIM / 128)      // prefetch next chunk's W2 panel
            __builtin_prefetch((const void*)(b2pan + 128), 0, 1);
          v16bf b = *(const v16bf*)(b2pan + k0);
          acc[t * 2 + 0] = __builtin_amdgcn_wmma_f32_16x16x32_bf16(
              false, a0, false, b, (short)0, acc[t * 2 + 0], false, false);
          acc[t * 2 + 1] = __builtin_amdgcn_wmma_f32_16x16x32_bf16(
              false, a1, false, b, (short)0, acc[t * 2 + 1], false, false);
        }
      }
      __syncthreads();   // done reading Hs before next chunk overwrites it
    }
  }

  // ------------- epilogue: add sum_e w_e*b2_e, store f32 tiles ------------
  #pragma unroll
  for (int t = 0; t < 8; ++t) {
    const int col = wave * 128 + t * 16 + nloc;
    float bcol[NEXP];
    #pragma unroll
    for (int e = 0; e < NEXP; ++e) bcol[e] = b2[(size_t)e * DDIM + col];
    #pragma unroll
    for (int m2 = 0; m2 < 2; ++m2) {
      #pragma unroll
      for (int r = 0; r < 8; ++r) {
        const int m = m2 * 16 + mbase + r;
        float bias = 0.f;
        #pragma unroll
        for (int e = 0; e < NEXP; ++e) bias += Wgate[m * NEXP + e] * bcol[e];
        out[(size_t)(t0 + m) * DDIM + col] = acc[t * 2 + m2][r] + bias;
      }
    }
  }
}

// -------------------------------------------------------------------
extern "C" void kernel_launch(void* const* d_in, const int* in_sizes, int n_in,
                              void* d_out, int out_size, void* d_ws, size_t ws_size,
                              hipStream_t stream) {
  const float* x  = (const float*)d_in[0];
  const float* Wg = (const float*)d_in[1];
  const float* bg = (const float*)d_in[2];
  const float* W1 = (const float*)d_in[3];
  const float* b1 = (const float*)d_in[4];
  const float* W2 = (const float*)d_in[5];
  const float* b2 = (const float*)d_in[6];
  float* out = (float*)d_out;

  char* ws = (char*)d_ws;
  float*  gates = (float*)(ws + OFF_GATES);
  __bf16* xb    = (__bf16*)(ws + OFF_XB);
  __bf16* W1b   = (__bf16*)(ws + OFF_W1B);
  __bf16* W2b   = (__bf16*)(ws + OFF_W2B);

  gate_kernel<<<N_TOK / 8, 256, 0, stream>>>(x, Wg, bg, gates);
  cvt_x_kernel<<<2048, 256, 0, stream>>>(x, xb);
  repack_w1_kernel<<<4096, 256, 0, stream>>>(W1, W1b);
  repack_w2_kernel<<<4096, 256, 0, stream>>>(W2, W2b);
  moe_main_kernel<<<N_TOK / 32, 256, 0, stream>>>(xb, W1b, W2b, b1, b2, gates, out);
}
